// ConvLSTM_6150393168656
// MI455X (gfx1250) — matrix operand order repaired and checked by reading, not compiled, not run on hardware
//
#include <hip/hip_runtime.h>
#include <hip/hip_bf16.h>

// ---------------------------------------------------------------------------
// Peephole LSTM (H=W=1 "ConvLSTM"): T=1024, B=128, CIN=256, HID=256.
//   * persistent kernel: 16 blocks (one 16-wide hid tile) x 8 waves (one
//     16-wide batch tile). Each wave computes all 4 gate tiles with
//     v_wmma_f32_16x16x32_bf16 (K=512 = concat(x_t, h_{t-1})); cell state c
//     lives in VGPRs for the whole sequence.
//   * Wk slice staged once in LDS (bf16, 528-elem padded rows).
//   * TDM double-buffer: tensor_load_to_lds streams next step's x slab
//     (128x256 bf16, padded to 528B rows) into LDS while the grid barrier
//     spins -> x loads leave the serial critical path.
//   * one grid-wide sense-reversing barrier per timestep publishes h (bf16).
// ---------------------------------------------------------------------------

typedef __attribute__((ext_vector_type(16))) __bf16 v16bf;
typedef __attribute__((ext_vector_type(8)))  __bf16 v8bf;
typedef __attribute__((ext_vector_type(8)))  float  v8f;
typedef __attribute__((ext_vector_type(4)))  unsigned u32x4;
typedef __attribute__((ext_vector_type(8)))  int      i32x8;
typedef __attribute__((ext_vector_type(4)))  int      i32x4;

#define T_STEPS 1024
#define BATCH   128
#define CIN     256
#define HIDC    256
#define KDIM    512            // CIN + HID
#define NBLK    16             // persistent blocks, one hid-tile each
#define TPB     256            // 8 waves of 32
#define LDSTRIDE 528           // Wk rows: 512 + 16 pad elems (1056B)
#define XROWB   528            // x slab row stride in bytes (512 + 16 pad)
#define BS_BYTES  (64 * LDSTRIDE * 2)   // 67584
#define XS_BYTES  (BATCH * XROWB)       // 67584

__device__ __forceinline__ v16bf make_frag(const __bf16* p0, const __bf16* p1) {
  union { v16bf v; v8bf p[2]; } u;
  u.p[0] = *(const v8bf*)p0;
  u.p[1] = *(const v8bf*)p1;
  return u.v;
}

__device__ __forceinline__ float sigmoidf_(float x) {
  return 1.f / (1.f + __expf(-x));
}
__device__ __forceinline__ float tanhf_(float x) {
  return 2.f / (1.f + __expf(-2.f * x)) - 1.f;
}

// TDM: load one [128 x 256] bf16 slab (row-major, contiguous) into LDS with
// 16B padding after each 512B row (pad_interval=128 dwords, pad_amount=4).
// This toolchain's builtin takes 6 args: (g0, g1, g2, g3, g4, cpol); our
// tensor is 2-D so groups 2..4 are zero/disabled.
__device__ __forceinline__ void tdm_load_x(const __bf16* gsrc, unsigned lds_addr) {
  unsigned long long ga = (unsigned long long)gsrc;
  u32x4 g0;
  g0[0] = 1u;                                       // count=1 (valid user D#)
  g0[1] = lds_addr;                                 // LDS byte address
  g0[2] = (unsigned)(ga & 0xffffffffull);           // global_addr[31:0]
  g0[3] = (unsigned)((ga >> 32) & 0x1ffffffull)     // global_addr[56:32]
        | (2u << 30);                               // type=2 ("image")
  i32x8 g1;
  g1[0] = (1 << 16)                                 // data_size = 2 bytes
        | (1 << 20)                                 // pad_enable
        | (6 << 22)                                 // pad_interval: 128 dwords
        | (3 << 25);                                // pad_amount: 4 dwords
  g1[1] = (int)(256u << 16);                        // tensor_dim0=256 (79:48)
  g1[2] = (int)(128u << 16);                        // tensor_dim1=128 (111:80)
  g1[3] = (int)(256u << 16);                        // tile_dim0=256 (127:112)
  g1[4] = 128;                                      // tile_dim1=128, tile_dim2=0
  g1[5] = 256;                                      // tensor_dim0_stride=256
  g1[6] = 0;
  g1[7] = 0;
  i32x4 z4 = {0, 0, 0, 0};
  i32x8 z8 = {0, 0, 0, 0, 0, 0, 0, 0};
#if __has_builtin(__builtin_amdgcn_tensor_load_to_lds)
  __builtin_amdgcn_tensor_load_to_lds(g0, g1, z4, z4, z8, 0);
#endif
}

// ---------------------------------------------------------------------------
__global__ void lstm_init(unsigned* cnt, unsigned* gen, __bf16* hbuf, int nh) {
  int i = blockIdx.x * blockDim.x + threadIdx.x;
  if (i == 0) { *cnt = 0u; *gen = 0u; }
  for (int j = i; j < nh; j += gridDim.x * blockDim.x)
    hbuf[j] = (__bf16)0.0f;
}

__global__ void cast_f32_bf16(const float* __restrict__ x,
                              __bf16* __restrict__ y, int n) {
  for (int i = blockIdx.x * blockDim.x + threadIdx.x; i < n;
       i += gridDim.x * blockDim.x)
    y[i] = (__bf16)x[i];
}

// ---------------------------------------------------------------------------
__global__ __launch_bounds__(TPB, 1) void lstm_persistent(
    const float* __restrict__ Wk,   // [1024][512] f32
    const float* __restrict__ bias, // [1024]
    const float* __restrict__ Wci,  // [256]
    const float* __restrict__ Wcf,  // [256]
    const float* __restrict__ Wco,  // [256]
    const __bf16* __restrict__ Xb,  // [T*B][256] bf16
    __bf16* __restrict__ hbuf,      // [B][256] bf16 recurrent state
    float* __restrict__ out,        // outs | h_last | c_last
    unsigned* cnt, unsigned* genp)
{
  // [Wk slice 67584B][x slab 0 67584B][x slab 1 67584B] = 198KB of 320KB
  __shared__ __attribute__((aligned(16))) unsigned char smem[BS_BYTES + 2 * XS_BYTES];
  __bf16* Bs = (__bf16*)smem;

  const int ht   = blockIdx.x;        // hid tile 0..15
  const int lane = threadIdx.x & 31;
  const int wave = threadIdx.x >> 5;  // batch tile 0..7
  const int r    = lane & 15;
  const int hh   = lane >> 4;

  const unsigned xs_lds[2] = { (unsigned)(uintptr_t)(smem + BS_BYTES),
                               (unsigned)(uintptr_t)(smem + BS_BYTES + XS_BYTES) };

  // Kick TDM for the t=0 x slab immediately; it overlaps the Wk staging below.
  if (wave == 0) tdm_load_x(Xb, xs_lds[0]);

  // Stage this block's Wk slice (4 gates x 16 rows x 512 K) into LDS as bf16.
  for (int idx = threadIdx.x; idx < 64 * KDIM; idx += TPB) {
    int row = idx >> 9;               // 0..63 = gate*16 + rr
    int k   = idx & (KDIM - 1);
    int g   = row >> 4, rr = row & 15;
    Bs[row * LDSTRIDE + k] = (__bf16)Wk[((g * HIDC) + ht * 16 + rr) * KDIM + k];
  }
  if (wave == 0) __builtin_amdgcn_s_wait_tensorcnt(0);
  __syncthreads();                    // Wk staged + x slab 0 resident

  const int hid = ht * 16 + r;        // this lane's hidden unit (N index)
  const float bi  = bias[0 * HIDC + hid];
  const float bff = bias[1 * HIDC + hid];
  const float bg  = bias[2 * HIDC + hid];
  const float bo  = bias[3 * HIDC + hid];
  const float wci = Wci[hid], wcf = Wcf[hid], wco = Wco[hid];

  v8f cst = {0.f, 0.f, 0.f, 0.f, 0.f, 0.f, 0.f, 0.f};  // cell state, resident

  const int arow  = wave * 16 + r;        // A-matrix row (batch) for this lane
  const int drow0 = wave * 16 + 8 * hh;   // first D-matrix row for this lane

  volatile unsigned* genv = genp;
  int cur = 0;

  for (int t = 0; t < T_STEPS; ++t) {
    // Prefetch next step's x slab into the other buffer; it is independent of
    // h, so the DMA overlaps this step's WMMAs and the grid barrier spin.
    if (wave == 0) {
      int tn = (t + 1 < T_STEPS) ? (t + 1) : t;
      tdm_load_x(Xb + (size_t)tn * BATCH * CIN, xs_lds[cur ^ 1]);
    }

    v8f acc0 = {bi,  bi,  bi,  bi,  bi,  bi,  bi,  bi };
    v8f acc1 = {bff, bff, bff, bff, bff, bff, bff, bff};
    v8f acc2 = {bg,  bg,  bg,  bg,  bg,  bg,  bg,  bg };
    v8f acc3 = {bo,  bo,  bo,  bo,  bo,  bo,  bo,  bo };

    const __bf16* xrow =
        (const __bf16*)(smem + BS_BYTES + cur * XS_BYTES + (size_t)arow * XROWB);
    const __bf16* hrow = hbuf + (size_t)arow * HIDC;

    // K = 0..255 : x part, A fragments from LDS (TDM-staged, padded rows).
#pragma unroll
    for (int kk = 0; kk < 8; ++kk) {
      const int k0 = kk * 32;
      v16bf a = make_frag(xrow + k0 + 8 * hh, xrow + k0 + 16 + 8 * hh);
      const __bf16* bb = Bs + r * LDSTRIDE + k0 + 8 * hh;
      v16bf b0 = make_frag(bb + 0 * 16 * LDSTRIDE, bb + 0 * 16 * LDSTRIDE + 16);
      v16bf b1 = make_frag(bb + 1 * 16 * LDSTRIDE, bb + 1 * 16 * LDSTRIDE + 16);
      v16bf b2 = make_frag(bb + 2 * 16 * LDSTRIDE, bb + 2 * 16 * LDSTRIDE + 16);
      v16bf b3 = make_frag(bb + 3 * 16 * LDSTRIDE, bb + 3 * 16 * LDSTRIDE + 16);
      acc0 = __builtin_amdgcn_wmma_f32_16x16x32_bf16(false, a, false, b0,
                                                     (short)0, acc0, false, false);
      acc1 = __builtin_amdgcn_wmma_f32_16x16x32_bf16(false, a, false, b1,
                                                     (short)0, acc1, false, false);
      acc2 = __builtin_amdgcn_wmma_f32_16x16x32_bf16(false, a, false, b2,
                                                     (short)0, acc2, false, false);
      acc3 = __builtin_amdgcn_wmma_f32_16x16x32_bf16(false, a, false, b3,
                                                     (short)0, acc3, false, false);
    }
    // K = 256..511 : h part, A fragments from the L2-resident global h buffer.
#pragma unroll
    for (int kk = 8; kk < 16; ++kk) {
      const int k0 = kk * 32;
      const __bf16* src = hrow + (k0 - CIN);
      v16bf a = make_frag(src + 8 * hh, src + 16 + 8 * hh);
      const __bf16* bb = Bs + r * LDSTRIDE + k0 + 8 * hh;
      v16bf b0 = make_frag(bb + 0 * 16 * LDSTRIDE, bb + 0 * 16 * LDSTRIDE + 16);
      v16bf b1 = make_frag(bb + 1 * 16 * LDSTRIDE, bb + 1 * 16 * LDSTRIDE + 16);
      v16bf b2 = make_frag(bb + 2 * 16 * LDSTRIDE, bb + 2 * 16 * LDSTRIDE + 16);
      v16bf b3 = make_frag(bb + 3 * 16 * LDSTRIDE, bb + 3 * 16 * LDSTRIDE + 16);
      acc0 = __builtin_amdgcn_wmma_f32_16x16x32_bf16(false, a, false, b0,
                                                     (short)0, acc0, false, false);
      acc1 = __builtin_amdgcn_wmma_f32_16x16x32_bf16(false, a, false, b1,
                                                     (short)0, acc1, false, false);
      acc2 = __builtin_amdgcn_wmma_f32_16x16x32_bf16(false, a, false, b2,
                                                     (short)0, acc2, false, false);
      acc3 = __builtin_amdgcn_wmma_f32_16x16x32_bf16(false, a, false, b3,
                                                     (short)0, acc3, false, false);
    }

    // Fused gates: lane owns (b = drow0+j, hid) for j=0..7; i,f,g,o all local.
    const size_t obase = ((size_t)t * BATCH + drow0) * HIDC + hid;
#pragma unroll
    for (int j = 0; j < 8; ++j) {
      float c  = cst[j];
      float iv = sigmoidf_(acc0[j] + wci * c);
      float fv = sigmoidf_(acc1[j] + wcf * c);
      c = fv * c + iv * tanhf_(acc2[j]);
      float ov = sigmoidf_(acc3[j] + wco * c);
      float hv = ov * tanhf_(c);
      cst[j] = c;
      out[obase + (size_t)j * HIDC] = hv;                       // outs[t]
      hbuf[(drow0 + j) * HIDC + hid] = (__bf16)hv;              // next-step A
      if (t == T_STEPS - 1) {
        size_t tail = (size_t)T_STEPS * BATCH * HIDC +
                      (size_t)(drow0 + j) * HIDC + hid;
        out[tail] = hv;                                         // h_last
        out[tail + (size_t)BATCH * HIDC] = c;                   // c_last
      }
    }

    // Publish h, then grid-wide sense-reversing barrier (16 arrivals).
    __threadfence();
    __syncthreads();
    if (threadIdx.x == 0) {
      unsigned g = *genv;
      if (atomicAdd(cnt, 1u) == NBLK - 1u) {
        *cnt = 0u;
        __threadfence();
        atomicAdd(genp, 1u);
      } else {
        while (*genv == g) { __builtin_amdgcn_s_sleep(2); }
      }
      __threadfence();
    }
    // Retire the TDM prefetch (issuing wave), then publish to the block.
    if (wave == 0) __builtin_amdgcn_s_wait_tensorcnt(0);
    __syncthreads();
    cur ^= 1;
  }
}

// ---------------------------------------------------------------------------
extern "C" void kernel_launch(void* const* d_in, const int* in_sizes, int n_in,
                              void* d_out, int out_size, void* d_ws, size_t ws_size,
                              hipStream_t stream) {
  const float* x   = (const float*)d_in[0];  // [T,B,256,1,1]
  const float* Wk  = (const float*)d_in[1];  // [1024,512]
  const float* b   = (const float*)d_in[2];  // [1024]
  const float* Wci = (const float*)d_in[3];  // [256]
  const float* Wcf = (const float*)d_in[4];  // [256]
  const float* Wco = (const float*)d_in[5];  // [256]
  float* out = (float*)d_out;

  // Workspace layout (~68 MB total):
  //   [0]    barrier count, [256B] barrier generation
  //   [4 KB] h state, bf16, B*HID = 64 KB
  //   [1 MB] X cast to bf16, T*B*CIN*2 = 64 MB
  unsigned* cnt  = (unsigned*)d_ws;
  unsigned* gen  = (unsigned*)((char*)d_ws + 256);
  __bf16*  hbuf  = (__bf16*)((char*)d_ws + 4096);
  __bf16*  xb    = (__bf16*)((char*)d_ws + (1u << 20));

  lstm_init<<<64, 256, 0, stream>>>(cnt, gen, hbuf, BATCH * HIDC);
  cast_f32_bf16<<<4096, 256, 0, stream>>>(x, xb, T_STEPS * BATCH * CIN);
  lstm_persistent<<<NBLK, TPB, 0, stream>>>(Wk, b, Wci, Wcf, Wco,
                                            xb, hbuf, out, cnt, gen);
}
